// CrossBundleAttention_68874095559260
// MI455X (gfx1250) — matrix operand order repaired and checked by reading, not compiled, not run on hardware
//
#include <hip/hip_runtime.h>
#include <hip/hip_bf16.h>
#include <cstdint>

// ---------------------------------------------------------------------------
// CrossBundleAttention for MI455X (gfx1250, wave32, WMMA).
//
// Every GEMM runs on v_wmma_f32_16x16x32_bf16 with fp32 accumulate. fp32
// operands are converted to bf16 in-register while building fragments (single
// streaming read of every weight byte, no conversion pass). Each wave computes
// a 32x64 output block (2x4 register tiles = 8 WMMAs per K-step) so A/B
// fragments are reused and L2 re-read traffic drops 4x/2x vs one-tile waves.
// A and B(NT) fragments use float4 (b128) loads along the contiguous K runs
// of the fragment k order used here (K = e + 16*hi).
// ---------------------------------------------------------------------------

typedef __attribute__((ext_vector_type(16))) __bf16 v16bf;
typedef __attribute__((ext_vector_type(8)))  float  v8f;

#define B_SZ   64
#define MEMLEN 2048
#define DLLM   4096
#define LMEM   16

// Load 8 consecutive fp32, convert to bf16 into dst[at..at+7].
__device__ __forceinline__ void ld_bf16x8(v16bf& dst, int at, const float* __restrict__ p)
{
    float4 x = *reinterpret_cast<const float4*>(p);
    float4 y = *reinterpret_cast<const float4*>(p + 4);
    dst[at + 0] = (__bf16)x.x; dst[at + 1] = (__bf16)x.y;
    dst[at + 2] = (__bf16)x.z; dst[at + 3] = (__bf16)x.w;
    dst[at + 4] = (__bf16)y.x; dst[at + 5] = (__bf16)y.y;
    dst[at + 6] = (__bf16)y.z; dst[at + 7] = (__bf16)y.w;
}

// A-matrix 16x32 fragment (k order used here, same as B): lane holds row (lane&15); element e
// maps to K = e + 16*hi  ->  one 16-wide contiguous run.
__device__ __forceinline__ void load_afrag(v16bf& a, const float* __restrict__ pA, int k0, int hi)
{
    ld_bf16x8(a, 0, pA + k0 + 16 * hi);
    ld_bf16x8(a, 8, pA + k0 + 16 * hi + 8);
}

// B (32x16) fragment from W^T where W is row-major (N x K): lane holds col
// (lane&15); element e maps to K = e + 16*hi -> one 16-wide contiguous run.
__device__ __forceinline__ void load_bfrag_nt(v16bf& b, const float* __restrict__ pW, int k0, int hi)
{
    ld_bf16x8(b, 0, pW + k0 + 16 * hi);
    ld_bf16x8(b, 8, pW + k0 + 16 * hi + 8);
}

// ---------------------------------------------------------------------------
// C[r, cOff + z*cHs + n] = scale * sum_k A[r, z*aHs + k] * W[n, z*wHs + k]
//                          (+ bias[z*bHs + n])
// One wave per 32x64 output block. grid = (N/64, M/32, nZ).
// ---------------------------------------------------------------------------
__global__ __launch_bounds__(32)
void gemm_nt_wmma(const float* __restrict__ A, int lda, int aHs,
                  const float* __restrict__ W, int ldw, int wHs,
                  const float* __restrict__ bias, int bHs,
                  float* __restrict__ C, int ldc, int cOff, int cHs,
                  int K, float scale)
{
    const int lane = threadIdx.x;
    const int hi   = lane >> 4;
    const int ln   = lane & 15;
    const int z    = blockIdx.z;
    const int m0   = blockIdx.y * 32;
    const int n0   = blockIdx.x * 64;

    const float* pA[2];
    pA[0] = A + (size_t)(m0 + ln) * lda + (size_t)z * aHs;
    pA[1] = pA[0] + (size_t)16 * lda;
    const float* pW[4];
#pragma unroll
    for (int j = 0; j < 4; ++j)
        pW[j] = W + (size_t)(n0 + 16 * j + ln) * ldw + (size_t)z * wHs;

    v8f acc[2][4] = {};
    for (int k0 = 0; k0 < K; k0 += 32) {
        if (k0 + 32 < K) {  // global_prefetch_b8
            __builtin_prefetch(pA[0] + k0 + 32, 0, 1);
            __builtin_prefetch(pW[0] + k0 + 32, 0, 1);
        }
        v16bf a[2], b[4];
#pragma unroll
        for (int i = 0; i < 2; ++i) load_afrag(a[i], pA[i], k0, hi);
#pragma unroll
        for (int j = 0; j < 4; ++j) load_bfrag_nt(b[j], pW[j], k0, hi);
#pragma unroll
        for (int i = 0; i < 2; ++i)
#pragma unroll
            for (int j = 0; j < 4; ++j)
                acc[i][j] = __builtin_amdgcn_wmma_f32_16x16x32_bf16(
                                false, a[i], false, b[j], (short)0, acc[i][j], false, false);
    }

#pragma unroll
    for (int i = 0; i < 2; ++i)
#pragma unroll
        for (int j = 0; j < 4; ++j) {
            const int col = n0 + 16 * j + ln;
            const float bv = bias ? bias[(size_t)z * bHs + col] : 0.0f;
#pragma unroll
            for (int v = 0; v < 8; ++v) {
                const int row = m0 + 16 * i + v + 8 * hi;
                C[(size_t)row * ldc + (size_t)cOff + (size_t)z * cHs + col] =
                    acc[i][j][v] * scale + bv;
            }
        }
}

// ---------------------------------------------------------------------------
// C[r, cOff + z*cHs + n] = scale * sum_k A[r, z*aHs + k] * Bm[k*ldb + z*bOff + n]
//                          (+ bias[z*bHs + n])
// One wave per 32x64 output block. grid = (N/64, M/32, nZ).
// ---------------------------------------------------------------------------
__global__ __launch_bounds__(32)
void gemm_nn_wmma(const float* __restrict__ A, int lda, int aHs,
                  const float* __restrict__ Bm, int ldb, long long bOff,
                  const float* __restrict__ bias, int bHs,
                  float* __restrict__ C, int ldc, int cOff, int cHs,
                  int K, float scale)
{
    const int lane = threadIdx.x;
    const int hi   = lane >> 4;
    const int ln   = lane & 15;
    const int z    = blockIdx.z;
    const int m0   = blockIdx.y * 32;
    const int n0   = blockIdx.x * 64;

    const float* pA[2];
    pA[0] = A + (size_t)(m0 + ln) * lda + (size_t)z * aHs;
    pA[1] = pA[0] + (size_t)16 * lda;
    const float* pB[4];
#pragma unroll
    for (int j = 0; j < 4; ++j)
        pB[j] = Bm + (size_t)z * (size_t)bOff + (n0 + 16 * j + ln);

    v8f acc[2][4] = {};
    for (int k0 = 0; k0 < K; k0 += 32) {
        if (k0 + 32 < K) {
            __builtin_prefetch(pA[0] + k0 + 32, 0, 1);
            __builtin_prefetch(pB[0] + (size_t)(k0 + 32) * ldb, 0, 1);
        }
        v16bf a[2], b[4];
#pragma unroll
        for (int i = 0; i < 2; ++i) load_afrag(a[i], pA[i], k0, hi);
#pragma unroll
        for (int j = 0; j < 4; ++j) {
#pragma unroll
            for (int e = 0; e < 16; ++e)
                b[j][e] = (__bf16)pB[j][(size_t)(k0 + e + 16 * hi) * ldb];
        }
#pragma unroll
        for (int i = 0; i < 2; ++i)
#pragma unroll
            for (int j = 0; j < 4; ++j)
                acc[i][j] = __builtin_amdgcn_wmma_f32_16x16x32_bf16(
                                false, a[i], false, b[j], (short)0, acc[i][j], false, false);
    }

#pragma unroll
    for (int i = 0; i < 2; ++i)
#pragma unroll
        for (int j = 0; j < 4; ++j) {
            const int col = n0 + 16 * j + ln;
            const float bv = bias ? bias[(size_t)z * bHs + col] : 0.0f;
#pragma unroll
            for (int v = 0; v < 8; ++v) {
                const int row = m0 + 16 * i + v + 8 * hi;
                C[(size_t)row * ldc + (size_t)cOff + (size_t)z * cHs + col] =
                    acc[i][j][v] * scale + bv;
            }
        }
}

// ---------------------------------------------------------------------------
// Masked softmax over M per (b, h) row. scores layout (B, nh, M).
// jnp bool mask -> 1 byte per element on device; True = ignore.
// ---------------------------------------------------------------------------
__global__ __launch_bounds__(256)
void softmax_mask_rows(float* __restrict__ sc, const unsigned char* __restrict__ mask,
                       int nh, int Mlen)
{
    __shared__ float red[256];
    const int t   = threadIdx.x;
    const int row = blockIdx.x;              // b*nh + h
    const int b   = row / nh;
    float* r = sc + (size_t)row * Mlen;
    const unsigned char* mk = mask + (size_t)b * Mlen;

    float mx = -3.0e38f;
    for (int m = t; m < Mlen; m += 256) {
        float v = mk[m] ? -1.0e9f : r[m];
        mx = fmaxf(mx, v);
    }
    red[t] = mx; __syncthreads();
    for (int s = 128; s > 0; s >>= 1) { if (t < s) red[t] = fmaxf(red[t], red[t + s]); __syncthreads(); }
    mx = red[0]; __syncthreads();

    float sum = 0.f;
    for (int m = t; m < Mlen; m += 256) {
        float v = mk[m] ? -1.0e9f : r[m];
        float e = __expf(v - mx);
        r[m] = e;
        sum += e;
    }
    red[t] = sum; __syncthreads();
    for (int s = 128; s > 0; s >>= 1) { if (t < s) red[t] += red[t + s]; __syncthreads(); }
    const float inv = 1.0f / red[0];
    for (int m = t; m < Mlen; m += 256) r[m] *= inv;
}

// ---------------------------------------------------------------------------
// LayerNorm over last dim D per row.
// ---------------------------------------------------------------------------
__global__ __launch_bounds__(256)
void layernorm_rows(const float* __restrict__ x, const float* __restrict__ g,
                    const float* __restrict__ b, float* __restrict__ out, int D)
{
    __shared__ float red[256];
    const int t = threadIdx.x;
    const float* r = x + (size_t)blockIdx.x * D;

    float s1 = 0.f;
    for (int i = t; i < D; i += 256) s1 += r[i];
    red[t] = s1; __syncthreads();
    for (int s = 128; s > 0; s >>= 1) { if (t < s) red[t] += red[t + s]; __syncthreads(); }
    const float mean = red[0] / D; __syncthreads();

    float s2 = 0.f;
    for (int i = t; i < D; i += 256) { float dv = r[i] - mean; s2 += dv * dv; }
    red[t] = s2; __syncthreads();
    for (int s = 128; s > 0; s >>= 1) { if (t < s) red[t] += red[t + s]; __syncthreads(); }
    const float rstd = rsqrtf(red[0] / D + 1e-5f);

    float* o = out + (size_t)blockIdx.x * D;
    for (int i = t; i < D; i += 256) o[i] = (r[i] - mean) * rstd * g[i] + b[i];
}

// ---------------------------------------------------------------------------
extern "C" void kernel_launch(void* const* d_in, const int* in_sizes, int n_in,
                              void* d_out, int out_size, void* d_ws, size_t ws_size,
                              hipStream_t stream)
{
    (void)in_sizes; (void)n_in; (void)out_size; (void)ws_size;

    const float* hidden = (const float*)d_in[0];
    const float* fib[3]  = {(const float*)d_in[1],  (const float*)d_in[2],  (const float*)d_in[3]};
    const unsigned char* mask = (const unsigned char*)d_in[4];
    const float* Wq[3]   = {(const float*)d_in[5],  (const float*)d_in[7],  (const float*)d_in[9]};
    const float* bq[3]   = {(const float*)d_in[6],  (const float*)d_in[8],  (const float*)d_in[10]};
    const float* Wqkv[3] = {(const float*)d_in[11], (const float*)d_in[15], (const float*)d_in[19]};
    const float* bqkv[3] = {(const float*)d_in[12], (const float*)d_in[16], (const float*)d_in[20]};
    const float* Wo[3]   = {(const float*)d_in[13], (const float*)d_in[17], (const float*)d_in[21]};
    const float* bo[3]   = {(const float*)d_in[14], (const float*)d_in[18], (const float*)d_in[22]};
    const float* Wl[3]   = {(const float*)d_in[23], (const float*)d_in[25], (const float*)d_in[27]};
    const float* bl[3]   = {(const float*)d_in[24], (const float*)d_in[26], (const float*)d_in[28]};
    const float* lng = (const float*)d_in[29];
    const float* lnb = (const float*)d_in[30];

    const int dF[3]       = {256, 512, 1024};
    const int nh[3]       = {4, 8, 16};
    const int S[3]        = {4, 8, 4};
    const int slotBase[3] = {0, 4, 12};

    // Workspace (fp32 elements), buffers sized for the largest fiber, reused.
    float* ws = (float*)d_ws;
    size_t off = 0;
    auto wsalloc = [&](size_t n) { float* p = ws + off; off += n; return p; };
    float* qf     = wsalloc((size_t)B_SZ * 1024);          // bundle query
    float* qp     = wsalloc((size_t)B_SZ * 1024);          // in-proj Q
    float* kp     = wsalloc((size_t)MEMLEN * 1024);        // in-proj K
    float* vp     = wsalloc((size_t)MEMLEN * 1024);        // in-proj V
    float* attn   = wsalloc((size_t)B_SZ * 16 * MEMLEN);   // scores / probs
    float* aout   = wsalloc((size_t)B_SZ * 1024);          // attn output (pre out-proj)
    float* ofib   = wsalloc((size_t)B_SZ * 1024);          // fiber output
    float* prefix = wsalloc((size_t)B_SZ * LMEM * DLLM);   // (B, 16, 4096) pre-LN

    const dim3 wave(32);

    for (int i = 0; i < 3; ++i) {
        const int d = dF[i], h = nh[i];

        // q_f = hidden @ Wq.T + bq                       (64 x d), K = 4096
        gemm_nt_wmma<<<dim3(d / 64, B_SZ / 32, 1), wave, 0, stream>>>(
            hidden, DLLM, 0, Wq[i], DLLM, 0, bq[i], 0,
            qf, d, 0, 0, DLLM, 1.0f);

        // kp = fiber @ Wk.T + bk                         (2048 x d), K = d
        gemm_nt_wmma<<<dim3(d / 64, MEMLEN / 32, 1), wave, 0, stream>>>(
            fib[i], d, 0, Wqkv[i] + (size_t)d * d, d, 0, bqkv[i] + d, 0,
            kp, d, 0, 0, d, 1.0f);

        // vp = fiber @ Wv.T + bv
        gemm_nt_wmma<<<dim3(d / 64, MEMLEN / 32, 1), wave, 0, stream>>>(
            fib[i], d, 0, Wqkv[i] + (size_t)2 * d * d, d, 0, bqkv[i] + 2 * d, 0,
            vp, d, 0, 0, d, 1.0f);

        // qp = q_f @ Wq_in.T + bq_in                     (64 x d), K = d
        gemm_nt_wmma<<<dim3(d / 64, B_SZ / 32, 1), wave, 0, stream>>>(
            qf, d, 0, Wqkv[i], d, 0, bqkv[i], 0,
            qp, d, 0, 0, d, 1.0f);

        // scores[b,z,m] = (qp_head . kp_head) / sqrt(64)  — per head z
        gemm_nt_wmma<<<dim3(MEMLEN / 64, B_SZ / 32, h), wave, 0, stream>>>(
            qp, d, 64, kp, d, 64, nullptr, 0,
            attn, h * MEMLEN, 0, MEMLEN, 64, 0.125f);

        // masked softmax over M per (b, head)
        softmax_mask_rows<<<dim3(B_SZ * h), dim3(256), 0, stream>>>(attn, mask, h, MEMLEN);

        // aout[b, z*64+n] = sum_m attn[b,z,m] * vp[m, z*64+n]
        gemm_nn_wmma<<<dim3(1, B_SZ / 32, h), wave, 0, stream>>>(
            attn, h * MEMLEN, MEMLEN, vp, d, 64, nullptr, 0,
            aout, d, 0, 64, MEMLEN, 1.0f);

        // o = aout @ Wo.T + bo
        gemm_nt_wmma<<<dim3(d / 64, B_SZ / 32, 1), wave, 0, stream>>>(
            aout, d, 0, Wo[i], d, 0, bo[i], 0,
            ofib, d, 0, 0, d, 1.0f);

        // slots[b, s, e] = o[b,:] @ Wlift[s] + blift[s]  -> prefix slab
        gemm_nn_wmma<<<dim3(DLLM / 64, B_SZ / 32, S[i]), wave, 0, stream>>>(
            ofib, d, 0, Wl[i], DLLM, (long long)d * DLLM, bl[i], DLLM,
            prefix, LMEM * DLLM, slotBase[i] * DLLM, DLLM, d, 1.0f);
    }

    // LayerNorm over d_LLM, (B*16) rows -> output
    layernorm_rows<<<dim3(B_SZ * LMEM), dim3(256), 0, stream>>>(
        prefix, lng, lnb, (float*)d_out, DLLM);
}
